// DCellLinear_36953898615445
// MI455X (gfx1250) — matrix operand order repaired and checked by reading, not compiled, not run on hardware
//
#include <hip/hip_runtime.h>
#include <stdint.h>

// Problem constants (match reference)
#define N_SUB  4096
#define BATCH  64
#define D_IN   128
#define D_OUT  128
#define ROW_DW 132   // padded fp32 LDS row stride in DWORDs (128 + 4, from TDM pad feature)
#define ROWB   136   // padded bf16 LDS row stride in elements (128 + 8)

typedef __attribute__((ext_vector_type(16))) __bf16   v16bf;
typedef __attribute__((ext_vector_type(8)))  __bf16   v8bf;
typedef __attribute__((ext_vector_type(4)))  __bf16   v4bf;
typedef __attribute__((ext_vector_type(8)))  float    v8f;
typedef __attribute__((ext_vector_type(4)))  uint32_t v4u;
typedef __attribute__((ext_vector_type(8)))  uint32_t v8u;

#define LDS_X_BYTES  (BATCH * ROW_DW * 4)            // 33792 B : X_s fp32 (TDM-staged)
#define LDS_W_BYTES  (D_OUT * ROW_DW * 4)            // 67584 B : W_s fp32 (TDM-staged)
#define LDS_XB_BYTES (BATCH * ROWB * 2)              // 17408 B : X_s bf16 (converted once)
#define LDS_TOTAL    (LDS_X_BYTES + LDS_W_BYTES + LDS_XB_BYTES)  // 118784 B < 320 KB/WGP

// LDS byte offset of a generic pointer known to alias LDS.
__device__ inline uint32_t lds_off(const void* p) {
    return (uint32_t)(uint64_t)(__attribute__((address_space(3))) const char*)p;
}

// Program one TDM 2-D descriptor and issue tensor_load_to_lds.
// Tile: rows x 128 fp32, contiguous in memory (row stride 128 elements).
// LDS pad: +4 DWORDs after every 128 DWORDs stored -> 132-DWORD LDS row stride,
// making all later ds_load_b128 fragment reads bank-conflict-free.
__device__ inline void tdm_load_2d_f32(uint32_t lds_addr, const void* gptr, uint32_t rows) {
    const uint64_t va = (uint64_t)(uintptr_t)gptr;
    v4u g0;
    g0[0] = 1u;                                   // count=1 (valid), user descriptor
    g0[1] = lds_addr;                             // lds_addr
    g0[2] = (uint32_t)va;                         // global_addr[31:0]
    g0[3] = (uint32_t)(va >> 32) | 0x80000000u;   // global_addr[56:32] | type=2
    v8u g1;
    g1[0] = (2u << 16)        // data_size = 2 -> 4-byte elements
          | (1u << 20)        // pad_enable
          | (6u << 22)        // pad_interval = 6 -> every 128 DWORDs (one 512B row)
          | (3u << 25);       // pad_amount   = 3 -> 4 DWORDs (16 B)
    g1[1] = (uint32_t)D_IN << 16;                 // tensor_dim0 = 128
    g1[2] = (rows & 0xFFFFu) << 16;               // tensor_dim1[15:0]
    g1[3] = (uint32_t)D_IN << 16;                 // tile_dim0 = 128
    g1[4] = rows;                                 // tile_dim1 = rows, tile_dim2 = 0
    g1[5] = (uint32_t)D_IN;                       // tensor_dim0_stride = 128
    g1[6] = 0u;
    g1[7] = 0u;
    // 2-D tensor: D# groups 2/3 NULL -> 2-operand form; inline asm is portable
    // across the 5-arg (ROCm 7.2) vs 6-arg (clang-23) builtin arity split.
    asm volatile("tensor_load_to_lds %0, %1" :: "s"(g0), "s"(g1) : "memory");
}

__device__ inline v8bf cvt8(float4 a, float4 b) {
    v8bf r;
    r[0] = (__bf16)a.x; r[1] = (__bf16)a.y; r[2] = (__bf16)a.z; r[3] = (__bf16)a.w;
    r[4] = (__bf16)b.x; r[5] = (__bf16)b.y; r[6] = (__bf16)b.z; r[7] = (__bf16)b.w;
    return r;
}

// One block per subsystem s; 256 threads = 8 wave32.
// TDM stages X_s and W_s into LDS; X converted to bf16 once; wave w owns
// output slab Y_s[:, 16w:16w+16] computed as 4 M-tiles x 4 K-chunk WMMAs.
__global__ __launch_bounds__(256)
void dcell_linear_wmma_tdm(const float* __restrict__ x,   // [N_SUB, BATCH, D_IN]
                           const float* __restrict__ W,   // [N_SUB, D_OUT, D_IN]
                           const float* __restrict__ b,   // [N_SUB, D_OUT]
                           float* __restrict__ y)         // [N_SUB*BATCH, D_OUT]
{
    extern __shared__ __align__(16) char smem[];
    float*  lx  = (float*)smem;                                  // [64][132]  fp32
    float*  lw  = (float*)(smem + LDS_X_BYTES);                  // [128][132] fp32
    __bf16* lxb = (__bf16*)(smem + LDS_X_BYTES + LDS_W_BYTES);   // [64][136]  bf16

    const int s    = blockIdx.x;
    const int t    = threadIdx.x;
    const int lane = t & 31;
    const int n    = lane & 15;                   // N (and A-row M) for this lane
    const int h    = lane >> 4;                   // lane half -> K selector
    const int wu   = __builtin_amdgcn_readfirstlane(t >> 5);  // wave id (SGPR)

    // ---- Async HBM->LDS staging via Tensor Data Mover (2 DMAs per wave) ----
    tdm_load_2d_f32(lds_off(lx) + (uint32_t)(wu * 8)  * (ROW_DW * 4),
                    x + ((size_t)s * BATCH + wu * 8)  * D_IN, 8u);
    tdm_load_2d_f32(lds_off(lw) + (uint32_t)(wu * 16) * (ROW_DW * 4),
                    W + ((size_t)s * D_OUT + wu * 16) * D_IN, 16u);
    __builtin_amdgcn_s_wait_tensorcnt(0);         // own DMAs complete
    __syncthreads();                              // publish all waves' tiles

    // ---- One-pass cooperative X fp32 -> bf16 conversion (done once per block,
    //      instead of 8x redundantly inside each wave's A-fragment assembly) ----
    #pragma unroll
    for (int j = 0; j < 8; ++j) {
        const int f   = t + 256 * j;              // logical float4 index 0..2047
        const int e   = f << 2;
        const int row = e >> 7;
        const int col = e & 127;
        const float4 v = *(const float4*)(lx + (size_t)row * ROW_DW + col);
        v4bf pv;
        pv[0] = (__bf16)v.x; pv[1] = (__bf16)v.y;
        pv[2] = (__bf16)v.z; pv[3] = (__bf16)v.w;
        *(v4bf*)(lxb + (size_t)row * ROWB + col) = pv;
    }
    __syncthreads();

    const int o = wu * 16 + n;                    // output feature / W row

    // ---- B fragments from LDS (fp32 -> bf16), register-resident for all 4 M-tiles ----
    // B(32x16): lane half h -> K = kbase+8h..+7 and kbase+16+8h..+7
    const float* wrow = lw + (size_t)o * ROW_DW;
    v16bf Bf[4];
    #pragma unroll
    for (int kc = 0; kc < 4; ++kc) {
        const float4* p = (const float4*)(wrow + kc * 32 + h * 8);
        Bf[kc] = __builtin_shufflevector(cvt8(p[0], p[1]), cvt8(p[4], p[5]),
                                         0,1,2,3,4,5,6,7,8,9,10,11,12,13,14,15);
    }

    const float bias = b[(size_t)s * D_OUT + o];

    #pragma unroll
    for (int mt = 0; mt < 4; ++mt) {
        v8f acc = {};
        const __bf16* xrow = lxb + (size_t)(mt * 16 + n) * ROWB;
        #pragma unroll
        for (int kc = 0; kc < 4; ++kc) {
            // A(16x32): lane half h -> K = kbase+8h..+7 and kbase+16+8h..+7
            const v8bf lo = *(const v8bf*)(xrow + kc * 32 + 8 * h);
            const v8bf hi = *(const v8bf*)(xrow + kc * 32 + 16 + 8 * h);
            const v16bf af = __builtin_shufflevector(lo, hi,
                              0,1,2,3,4,5,6,7,8,9,10,11,12,13,14,15);
            acc = __builtin_amdgcn_wmma_f32_16x16x32_bf16(
                      false, af, false, Bf[kc], (short)0, acc, false, false);
        }
        // D layout: lane -> col n, rows h*8 + r
        float* yp = y + ((size_t)s * BATCH + mt * 16 + h * 8) * D_OUT + o;
        #pragma unroll
        for (int r = 0; r < 8; ++r)
            yp[(size_t)r * D_OUT] = acc[r] + bias;
    }
}

extern "C" void kernel_launch(void* const* d_in, const int* in_sizes, int n_in,
                              void* d_out, int out_size, void* d_ws, size_t ws_size,
                              hipStream_t stream) {
    const float* x = (const float*)d_in[0];
    const float* W = (const float*)d_in[1];
    const float* b = (const float*)d_in[2];
    float*       y = (float*)d_out;
    (void)in_sizes; (void)n_in; (void)out_size; (void)d_ws; (void)ws_size;

    dcell_linear_wmma_tdm<<<dim3(N_SUB), dim3(256), LDS_TOTAL, stream>>>(x, W, b, y);
}
